// QuantizedRNN_84447646974236
// MI455X (gfx1250) — matrix operand order repaired and checked by reading, not compiled, not run on hardware
//
#include <hip/hip_runtime.h>
#include <hip/hip_bf16.h>

// ---------------------------------------------------------------------------
// QuantizedRNN (2-layer tanh RNN) for gfx1250 / MI455X.
// quantize(E5M10, RNE) == fp16 cast -> v_wmma_f32_16x16x32_f16 reproduces the
// reference quantization of x/W exactly (f32 accumulate).
// Persistent kernel: 16 WGs x 256 thr = 128 waves = one 16x16 tile each
// (M=32 -> 2 tiles, N=1024 -> 64 tiles). Weights pre-packed into WMMA-B
// layout, L2-resident (7MB << 192MB). Two grid barriers per time step.
// A-matrix staged into LDS with GLOBAL_LOAD_ASYNC_TO_LDS_B128 when available.
// ---------------------------------------------------------------------------

typedef __attribute__((ext_vector_type(16))) _Float16 v16h;
typedef __attribute__((ext_vector_type(8)))  _Float16 v8h;
typedef __attribute__((ext_vector_type(4)))  _Float16 v4h;
typedef __attribute__((ext_vector_type(8)))  float    v8f;
typedef __attribute__((ext_vector_type(4)))  int      v4i;
typedef v4i __attribute__((address_space(1))) v4i_glob;
typedef v4i __attribute__((address_space(3))) v4i_lds;

#define BATCHN 32
#define SEQT   1024
#define DIN    512
#define HID    1024
#define K0     (DIN + HID)    // 1536  (layer0: [x | h0])
#define K1     (HID + HID)    // 2048  (layer1: [h0 | h1])
#define NWG    16
#define LDSW   (K1 + 24)      // halfs per LDS row; +24 halfs kills bank conflicts
#define LDS_BYTES (BATCHN * LDSW * 2)

__device__ __forceinline__ float fast_tanh(float x) {
#if __has_builtin(__builtin_amdgcn_tanhf)
  return __builtin_amdgcn_tanhf(x);          // v_tanh_f32 (native TRANS op)
#else
  return tanhf(x);
#endif
}

// ---- async global->LDS staging (gfx1250 GLOBAL_LOAD_ASYNC_TO_LDS_B128) ----
#if __has_builtin(__builtin_amdgcn_global_load_async_to_lds_b128)
#define HAVE_ASYNC_LDS 1
__device__ __forceinline__ void copy16_to_lds(_Float16* lds_dst,
                                              const _Float16* gsrc) {
  __builtin_amdgcn_global_load_async_to_lds_b128(
      (v4i_glob*)gsrc, (v4i_lds*)lds_dst, 0, 0);
}
__device__ __forceinline__ void wait_async_lds() {
#if __has_builtin(__builtin_amdgcn_s_wait_asynccnt)
  __builtin_amdgcn_s_wait_asynccnt(0);
#else
  asm volatile("s_wait_asynccnt 0" ::: "memory");
#endif
}
#else
#define HAVE_ASYNC_LDS 0
__device__ __forceinline__ void copy16_to_lds(_Float16* lds_dst,
                                              const _Float16* gsrc) {
  *(v8h*)lds_dst = *(const v8h*)gsrc;        // global_load_b128 + ds_store_b128
}
__device__ __forceinline__ void wait_async_lds() {}
#endif

// Device-wide barrier (persistent grid, counters live in d_ws, re-zeroed
// every call by init_kernel -> deterministic).
__device__ __forceinline__ void grid_sync(unsigned* cnt, unsigned* gen, unsigned nwg) {
  __syncthreads();
  if (threadIdx.x == 0) {
    __threadfence();
    volatile unsigned* vgen = (volatile unsigned*)gen;
    unsigned g = *vgen;
    unsigned arrived = atomicAdd(cnt, 1u) + 1u;
    if (arrived == nwg) {
      atomicExch(cnt, 0u);
      __threadfence();
      atomicAdd(gen, 1u);
    } else {
      while (*vgen == g) { __builtin_amdgcn_s_sleep(1); }
    }
    __threadfence();
  }
  __syncthreads();
}

// One 16x16 output tile: C = bias + A(16xK, LDS) * B(Kx16, packed global).
// A fragment (ISA 7.12.2, 16-bit A 16x32): lane 0-15 -> M=lane, halves 0-7 =
// K+0..7, halves 8-15 = K+16..23; lanes 16-31 -> K+8..15 / K+24..31.
// B packed [n][k] per tile: lane reads column n=lane&15, same K order as A.
__device__ __forceinline__ v8f tile_gemm(const _Float16* __restrict__ As,
                                         int arow0, int ktot,
                                         const _Float16* __restrict__ Bt,
                                         float bz, int lane) {
  v8f acc;
#pragma unroll
  for (int i = 0; i < 8; ++i) acc[i] = bz;
  const int m  = arow0 + (lane & 15);
  const int ko = (lane >> 4) << 3;                 // 0 or 8
  const _Float16* ap = As + m * LDSW + ko;
  const _Float16* bp = Bt + (size_t)(lane & 15) * ktot + ko;   // lane's N-column
  union AB { v16h v; v8h h[2]; };
#pragma unroll 4
  for (int kb = 0; kb < ktot; kb += 32) {
    AB a, b;
    a.h[0] = *(const v8h*)(ap + kb);          // ds_load_b128
    a.h[1] = *(const v8h*)(ap + kb + 16);     // ds_load_b128
    b.h[0] = *(const v8h*)(bp + kb);          // global_load_b128
    b.h[1] = *(const v8h*)(bp + kb + 16);     // global_load_b128
    acc = __builtin_amdgcn_wmma_f32_16x16x32_f16(
        false, a.v, false, b.v, (short)0, acc, false, false);
  }
  return acc;
}

// ---------------------------------------------------------------------------
// Prep kernels
// ---------------------------------------------------------------------------
__global__ void quantize_x_kernel(const float* __restrict__ x,
                                  _Float16* __restrict__ qx, int n4) {
  int i = blockIdx.x * blockDim.x + threadIdx.x;
  if (i < n4) {
    const float4 v = ((const float4*)x)[i];
    v4h o;
    o[0] = (_Float16)v.x; o[1] = (_Float16)v.y;
    o[2] = (_Float16)v.z; o[3] = (_Float16)v.w;
    *(v4h*)&qx[(size_t)i * 4] = o;            // RNE f32->f16 == quantize()
  }
}

// Pack [W_ih | W_hh] (both [N=1024][K]) into WMMA-B tiles:
// Bp[((n16*16 + j) * Kt) + k] = (half)W[n16*16+j][k]
__global__ void pack_w_kernel(const float* __restrict__ wih,
                              const float* __restrict__ whh,
                              _Float16* __restrict__ Bp, int kih, int khh) {
  const int kt = kih + khh;
  const size_t total = (size_t)(HID / 16) * kt * 16;
  for (size_t i = (size_t)blockIdx.x * blockDim.x + threadIdx.x; i < total;
       i += (size_t)gridDim.x * blockDim.x) {
    int k = (int)(i % kt);
    size_t r = i / kt;
    int j   = (int)(r & 15);
    int n16 = (int)(r >> 4);
    int nn  = n16 * 16 + j;
    float w = (k < kih) ? wih[(size_t)nn * kih + k]
                        : whh[(size_t)nn * khh + (k - kih)];
    Bp[i] = (_Float16)w;
  }
}

__global__ void bias_kernel(const float* bi0, const float* bh0, float* o0,
                            const float* bi1, const float* bh1, float* o1) {
  int i = blockIdx.x * blockDim.x + threadIdx.x;
  if (i < HID) {
    o0[i] = (float)(_Float16)bi0[i] + (float)(_Float16)bh0[i];
    o1[i] = (float)(_Float16)bi1[i] + (float)(_Float16)bh1[i];
  }
}

__global__ void init_kernel(unsigned* __restrict__ sync,
                            _Float16* __restrict__ h0,
                            _Float16* __restrict__ h1) {
  int i = blockIdx.x * blockDim.x + threadIdx.x;
  if (i < 64) sync[i] = 0u;
  for (int c = i; c < 2 * BATCHN * HID; c += gridDim.x * blockDim.x) {
    h0[c] = (_Float16)0.0f;
    h1[c] = (_Float16)0.0f;
  }
}

// ---------------------------------------------------------------------------
// Persistent recurrent kernel
// ---------------------------------------------------------------------------
__global__ void __launch_bounds__(256, 1)
rnn_persistent_kernel(const _Float16* __restrict__ qx,
                      const _Float16* __restrict__ B0,
                      const _Float16* __restrict__ B1,
                      const float* __restrict__ bias0,
                      const float* __restrict__ bias1,
                      _Float16* __restrict__ h0buf,   // 2 x [32][1024] f16
                      _Float16* __restrict__ h1buf,   // 2 x [32][1024] f16
                      float* __restrict__ out,        // [32][1024][1024] f32 + hfinal
                      unsigned* __restrict__ sync) {
  extern __shared__ _Float16 As[];                    // [32][LDSW]
  const int tid  = threadIdx.x;
  const int lane = tid & 31;
  const int gw   = blockIdx.x * (blockDim.x >> 5) + (tid >> 5);  // 0..127
  const int nt   = gw & 63;
  const int mt   = gw >> 6;
  const int n    = nt * 16 + (lane & 15);             // C/D: N = lane&15
  const int mrow = (lane >> 4) * 8;                   // C/D: M = hi*8 + r
  float* hfinal  = out + (size_t)BATCHN * SEQT * HID;
  unsigned* cnt = sync;
  unsigned* gen = sync + 1;
  const float bz0 = bias0[n];
  const float bz1 = bias1[n];
  const _Float16* Bt0 = B0 + (size_t)nt * K0 * 16;
  const _Float16* Bt1 = B1 + (size_t)nt * K1 * 16;

  for (int t = 0; t < SEQT; ++t) {
    const int rb = t & 1, wb = rb ^ 1;

    // ----- layer 0: A = [qx_t (32x512) | h0_prev (32x1024)] -----
    for (int c = tid; c < (BATCHN * DIN) / 8; c += 256) {   // 2048 chunks
      int m = c >> 6, kc = (c & 63) << 3;
      copy16_to_lds(&As[m * LDSW + kc],
                    &qx[((size_t)m * SEQT + t) * DIN + kc]);
    }
    {
      const _Float16* h0r = h0buf + rb * (BATCHN * HID);
      for (int c = tid; c < (BATCHN * HID) / 8; c += 256) { // 4096 chunks
        int m = c >> 7, kc = (c & 127) << 3;
        copy16_to_lds(&As[m * LDSW + DIN + kc], &h0r[m * HID + kc]);
      }
    }
    wait_async_lds();
    __syncthreads();

    v8f acc0 = tile_gemm(As, mt * 16, K0, Bt0, bz0, lane);

    _Float16* h0w = h0buf + wb * (BATCHN * HID);
    {
      float hv[8];
#pragma unroll
      for (int r = 0; r < 8; ++r) hv[r] = fast_tanh(acc0[r]);
#pragma unroll
      for (int r = 0; r < 8; ++r)
        h0w[(mt * 16 + mrow + r) * HID + n] = (_Float16)hv[r];
      if (t == SEQT - 1) {
#pragma unroll
        for (int r = 0; r < 8; ++r)
          hfinal[(mt * 16 + mrow + r) * HID + n] = hv[r];   // h_final layer 0
      }
    }
    grid_sync(cnt, gen, NWG);

    // ----- layer 1: A = [h0_new (32x1024) | h1_prev (32x1024)] -----
    {
      const _Float16* h0n = h0buf + wb * (BATCHN * HID);
      const _Float16* h1r = h1buf + rb * (BATCHN * HID);
      for (int c = tid; c < (BATCHN * HID) / 8; c += 256) {
        int m = c >> 7, kc = (c & 127) << 3;
        copy16_to_lds(&As[m * LDSW + kc], &h0n[m * HID + kc]);
      }
      for (int c = tid; c < (BATCHN * HID) / 8; c += 256) {
        int m = c >> 7, kc = (c & 127) << 3;
        copy16_to_lds(&As[m * LDSW + HID + kc], &h1r[m * HID + kc]);
      }
    }
    wait_async_lds();
    __syncthreads();

    v8f acc1 = tile_gemm(As, mt * 16, K1, Bt1, bz1, lane);

    _Float16* h1w = h1buf + wb * (BATCHN * HID);
    {
      float hv[8];
#pragma unroll
      for (int r = 0; r < 8; ++r) hv[r] = fast_tanh(acc1[r]);
#pragma unroll
      for (int r = 0; r < 8; ++r) {
        int b = mt * 16 + mrow + r;
        h1w[b * HID + n] = (_Float16)hv[r];
        out[((size_t)b * SEQT + t) * HID + n] = hv[r];      // y[b,t,n]
      }
      if (t == SEQT - 1) {
#pragma unroll
        for (int r = 0; r < 8; ++r)
          hfinal[BATCHN * HID + (mt * 16 + mrow + r) * HID + n] = hv[r];
      }
    }
    grid_sync(cnt, gen, NWG);
  }
}

// ---------------------------------------------------------------------------
// Host entry
// ---------------------------------------------------------------------------
extern "C" void kernel_launch(void* const* d_in, const int* in_sizes, int n_in,
                              void* d_out, int out_size, void* d_ws, size_t ws_size,
                              hipStream_t stream) {
  (void)in_sizes; (void)n_in; (void)out_size; (void)ws_size;
  const float* x     = (const float*)d_in[0];
  const float* w_ih0 = (const float*)d_in[1];
  const float* w_hh0 = (const float*)d_in[2];
  const float* b_ih0 = (const float*)d_in[3];
  const float* b_hh0 = (const float*)d_in[4];
  const float* w_ih1 = (const float*)d_in[5];
  const float* w_hh1 = (const float*)d_in[6];
  const float* b_ih1 = (const float*)d_in[7];
  const float* b_hh1 = (const float*)d_in[8];
  float* out = (float*)d_out;

  // Workspace carve-up (~41.2 MB total)
  char* ws = (char*)d_ws;
  size_t off = 0;
  unsigned* sync = (unsigned*)(ws + off); off += 256;
  _Float16* qx   = (_Float16*)(ws + off); off += (size_t)BATCHN * SEQT * DIN * 2;
  _Float16* B0   = (_Float16*)(ws + off); off += (size_t)(HID / 16) * K0 * 16 * 2;
  _Float16* B1   = (_Float16*)(ws + off); off += (size_t)(HID / 16) * K1 * 16 * 2;
  float* bias0   = (float*)(ws + off);    off += HID * 4;
  float* bias1   = (float*)(ws + off);    off += HID * 4;
  _Float16* h0b  = (_Float16*)(ws + off); off += 2 * BATCHN * HID * 2;
  _Float16* h1b  = (_Float16*)(ws + off); off += 2 * BATCHN * HID * 2;

  // Allow >64KB dynamic LDS (no-op if unsupported; not a stream op).
  (void)hipFuncSetAttribute((const void*)rnn_persistent_kernel,
                            hipFuncAttributeMaxDynamicSharedMemorySize,
                            LDS_BYTES);

  init_kernel<<<64, 256, 0, stream>>>(sync, h0b, h1b);

  const int n4 = (BATCHN * SEQT * DIN) / 4;
  quantize_x_kernel<<<(n4 + 255) / 256, 256, 0, stream>>>(x, qx, n4);

  pack_w_kernel<<<2048, 256, 0, stream>>>(w_ih0, w_hh0, B0, DIN, HID);
  pack_w_kernel<<<2048, 256, 0, stream>>>(w_ih1, w_hh1, B1, HID, HID);
  bias_kernel<<<(HID + 255) / 256, 256, 0, stream>>>(b_ih0, b_hh0, bias0,
                                                     b_ih1, b_hh1, bias1);

  rnn_persistent_kernel<<<NWG, 256, LDS_BYTES, stream>>>(
      qx, B0, B1, bias0, bias1, h0b, h1b, out, sync);
}
